// GenVar_90890097918452
// MI455X (gfx1250) — matrix operand order repaired and checked by reading, not compiled, not run on hardware
//
#include <hip/hip_runtime.h>
#include <hip/hip_bf16.h>
#include <math.h>

typedef __bf16 v16bf __attribute__((ext_vector_type(16)));
typedef __bf16 v8bf  __attribute__((ext_vector_type(8)));
typedef __bf16 v2bf  __attribute__((ext_vector_type(2)));
typedef float  v8f   __attribute__((ext_vector_type(8)));

#define B_   32
#define S_   4096
#define DV_  512
#define DQ_  512
#define H_   256

#define CHUNK  256            // S rows handled per workgroup
#define NCHUNK (S_/CHUNK)     // 16 chunks per batch
#define TILE   16
#define NTILE  (CHUNK/TILE)   // 16 s-tiles per chunk

#define W1T_STRIDE  528       // bf16/row: 512 + 16 pad (rows 16B aligned, banks spread by 8)
#define CTXB_STRIDE 520       // bf16/row: 512 + 8 pad (rows 16B aligned, banks spread by 4)
#define CTX_STRIDE  516       // f32/row: 512 + 4 pad

// dynamic LDS layout (bytes)
#define OFF_W1T  0
#define SZ_W1T   (H_*W1T_STRIDE*2)             // 270336
#define OFF_CTXB (OFF_W1T + SZ_W1T)
#define SZ_CTXB  (TILE*CTXB_STRIDE*2)          // 16640
#define OFF_CTX  (OFF_CTXB + SZ_CTXB)          // 286976 (16B aligned)
#define SZ_CTX   (TILE*CTX_STRIDE*4)           // 33024
#define OFF_QB   (OFF_CTX + SZ_CTX)
#define OFF_WV   (OFF_QB + H_*4)
#define OFF_WS   (OFF_WV + H_*4)               // per-wave score partials: 8 waves x 16 rows
#define OFF_E    (OFF_WS + 8*TILE*4)
#define SMEM_BYTES (OFF_E + TILE*4)            // 322624 B <= 327680 B (320 KB WGP LDS)

// ---------------- kernel 1: qb[b,h] = query[b,:] @ W2[:,h] + bias[h] ----------------
__global__ void qb_kernel(const float* __restrict__ q, const float* __restrict__ W2,
                          const float* __restrict__ bias, float* __restrict__ qb) {
    int b = blockIdx.x, h = threadIdx.x;          // grid B_, block H_
    float acc = bias[h];
    for (int d = 0; d < DQ_; ++d)
        acc = fmaf(q[b*DQ_ + d], W2[d*H_ + h], acc);
    qb[b*H_ + h] = acc;
}

// ---------------- kernel 2: fused GEMM + tanh + dot(w) + exp + weighted ctx ----------------
__global__ __launch_bounds__(256)
void attn_chunk_kernel(const float* __restrict__ ctx, const int* __restrict__ lens,
                       const float* __restrict__ W1, const float* __restrict__ qb,
                       const float* __restrict__ wvec, float* __restrict__ partial) {
    extern __shared__ char smem[];
    __bf16* sW1T  = (__bf16*)(smem + OFF_W1T);   // [H_][W1T_STRIDE]  W1 transposed, bf16
    __bf16* sctxb = (__bf16*)(smem + OFF_CTXB);  // [TILE][CTXB_STRIDE] ctx tile, bf16 (WMMA A feed)
    float*  sctx  = (float*) (smem + OFF_CTX);   // [TILE][CTX_STRIDE]  ctx tile, f32 (weighted sum)
    float*  sqb   = (float*) (smem + OFF_QB);
    float*  sw    = (float*) (smem + OFF_WV);
    float*  swsc  = (float*) (smem + OFF_WS);    // [8][16] per-wave score partials
    float*  se    = (float*) (smem + OFF_E);     // [16] exp(score)

    const int tid   = threadIdx.x;
    const int chunk = blockIdx.x;
    const int b     = blockIdx.y;
    const int lane  = tid & 31;
    const int wave  = tid >> 5;
    const int hi    = lane >> 4;      // K-half selector in WMMA A/B layouts
    const int ln    = lane & 15;
    const int len   = lens[b];

    // stage W1 as bf16, transposed; packed dword stores (both source rows coalesced)
    for (int pair = tid; pair < (DV_/2)*H_; pair += 256) {
        int k2  = pair >> 8;          // k pair index (H_ == 256)
        int col = pair & 255;
        v2bf pw;
        pw[0] = (__bf16)W1[(2*k2    )*H_ + col];
        pw[1] = (__bf16)W1[(2*k2 + 1)*H_ + col];
        *(v2bf*)&sW1T[col*W1T_STRIDE + 2*k2] = pw;
    }
    if (tid < H_) { sqb[tid] = qb[b*H_ + tid]; sw[tid] = wvec[tid]; }
    __syncthreads();

    float2 accv = make_float2(0.f, 0.f);
    float  esum = 0.f;
    const int    s0    = chunk * CHUNK;
    const float* ctx_b = ctx + (size_t)b * S_ * DV_;
    const int    nt0   = wave * 2, nt1 = wave * 2 + 1;   // each wave owns 2 N-tiles
    const int    c0    = tid * 2;                        // 2 DV columns per thread

    for (int t = 0; t < NTILE; ++t) {
        const int srow = s0 + t * TILE;
        // ---- load 16x512 f32 tile (coalesced float2); convert to bf16 once ----
        const float2* g = (const float2*)(ctx_b + (size_t)srow * DV_);
        #pragma unroll
        for (int m = 0; m < TILE; ++m) {
            float2 v = g[m*256 + tid];
            *(float2*)(sctx + m*CTX_STRIDE + 2*tid) = v;
            v2bf pv; pv[0] = (__bf16)v.x; pv[1] = (__bf16)v.y;
            *(v2bf*)&sctxb[m*CTXB_STRIDE + 2*tid] = pv;
        }
        __syncthreads();

        // prefetch next tile into cache hierarchy (gfx1250 global_prefetch path)
        if (t + 1 < NTILE) {
            const char* np = (const char*)(ctx_b + (size_t)(srow + TILE) * DV_);
            __builtin_prefetch(np + tid * 128, 0, 3);   // 256 thr * 128B = 32KB tile
        }

        // ---- WMMA: (16 x 512) bf16 @ (512 x 32) bf16 -> 16 x 32 f32 per wave ----
        v8f acc0 = {}; v8f acc1 = {};
        const __bf16* arow = &sctxb[ln * CTXB_STRIDE];
        #pragma unroll
        for (int kb = 0; kb < 16; ++kb) {
            // A fragment 16x32 bf16: lane (hi,ln): elem j -> K = kb*32 + hi*16 + j
            // => two aligned 16B LDS loads at dword offsets {hi*8, hi*8+4} within the kb block
            v8bf alo = *(const v8bf*)&arow[kb*32 + hi*16];
            v8bf ahi = *(const v8bf*)&arow[kb*32 + hi*16 + 8];
            v16bf a = __builtin_shufflevector(alo, ahi,
                        0,1,2,3,4,5,6,7, 8,9,10,11,12,13,14,15);
            // B fragment 32x16 bf16: lane (hi,ln): col = nt*16+ln, elem j -> K = kb*32 + hi*16 + j
            const v16bf* bp0 = (const v16bf*)&sW1T[(nt0*16 + ln)*W1T_STRIDE + kb*32 + hi*16];
            const v16bf* bp1 = (const v16bf*)&sW1T[(nt1*16 + ln)*W1T_STRIDE + kb*32 + hi*16];
            acc0 = __builtin_amdgcn_wmma_f32_16x16x32_bf16(false, a, false, *bp0,
                                                           (short)0, acc0, false, false);
            acc1 = __builtin_amdgcn_wmma_f32_16x16x32_bf16(false, a, false, *bp1,
                                                           (short)0, acc1, false, false);
        }

        // ---- scores: tanh(u + qb) . w, reduced over columns ----
        // D layout: lanes 0-15 -> M = r, lanes 16-31 -> M = 8 + r ; N = ln
        const int col0 = nt0*16 + ln, col1 = nt1*16 + ln;
        const float qb0 = sqb[col0], w0 = sw[col0];
        const float qb1 = sqb[col1], w1 = sw[col1];
        float rs[8];
        #pragma unroll
        for (int r = 0; r < 8; ++r)
            rs[r] = tanhf(acc0[r] + qb0) * w0 + tanhf(acc1[r] + qb1) * w1;
        #pragma unroll
        for (int r = 0; r < 8; ++r) {           // butterfly within each 16-lane half
            float v = rs[r];
            v += __shfl_xor(v, 1);
            v += __shfl_xor(v, 2);
            v += __shfl_xor(v, 4);
            v += __shfl_xor(v, 8);
            rs[r] = v;
        }
        if (ln == 0) {
            #pragma unroll
            for (int r = 0; r < 8; ++r)
                swsc[wave*16 + hi*8 + r] = rs[r];   // unique slots, deterministic
        }
        __syncthreads();

        if (tid < TILE) {                       // fixed-order cross-wave reduction
            float sc = 0.f;
            #pragma unroll
            for (int w = 0; w < 8; ++w) sc += swsc[w*16 + tid];
            const int sidx = srow + tid;
            se[tid] = (sidx < len) ? __expf(sc) : 0.f;   // tanh-bounded: no max-sub needed
        }
        __syncthreads();

        // ---- online accumulation: esum += e, accv += e * ctx_row (f32 tile) ----
        #pragma unroll
        for (int m = 0; m < TILE; ++m) {
            const float e = se[m];
            esum   += e;
            accv.x += e * sctx[m*CTX_STRIDE + c0];
            accv.y += e * sctx[m*CTX_STRIDE + c0 + 1];
        }
        __syncthreads();   // before next tile overwrites sctx/sctxb
    }

    float* P = partial + (size_t)(b*NCHUNK + chunk) * (DV_ + 1);
    P[c0]     = accv.x;
    P[c0 + 1] = accv.y;
    if (tid == 0) P[DV_] = esum;
}

// ---------------- kernel 3: reduce chunks, normalize ----------------
__global__ void finalize_kernel(const float* __restrict__ partial, float* __restrict__ out) {
    int b = blockIdx.x, dv = threadIdx.x;       // grid B_, block DV_
    float acc = 0.f, es = 0.f;
    for (int c = 0; c < NCHUNK; ++c) {
        const float* P = partial + (size_t)(b*NCHUNK + c) * (DV_ + 1);
        acc += P[dv];
        es  += P[DV_];
    }
    out[b*DV_ + dv] = acc / es;
}

extern "C" void kernel_launch(void* const* d_in, const int* in_sizes, int n_in,
                              void* d_out, int out_size, void* d_ws, size_t ws_size,
                              hipStream_t stream) {
    (void)in_sizes; (void)n_in; (void)out_size; (void)ws_size;
    const float* enc  = (const float*)d_in[0];  // (1,B,DQ)
    const float* ctx  = (const float*)d_in[1];  // (B,S,DV)
    const int*   lens = (const int*)  d_in[2];  // (B,)
    const float* W1   = (const float*)d_in[3];  // (DV,H)
    const float* W2   = (const float*)d_in[4];  // (DQ,H)
    const float* bias = (const float*)d_in[5];  // (1,1,H)
    const float* wv   = (const float*)d_in[6];  // (H,1)
    float* out = (float*)d_out;                 // (B,DV)

    float* qb      = (float*)d_ws;              // B_*H_ floats
    float* partial = qb + B_*H_;                // B_*NCHUNK*(DV_+1) floats

    hipFuncSetAttribute((const void*)attn_chunk_kernel,
                        hipFuncAttributeMaxDynamicSharedMemorySize, SMEM_BYTES);

    qb_kernel<<<B_, H_, 0, stream>>>(enc, W2, bias, qb);
    attn_chunk_kernel<<<dim3(NCHUNK, B_), 256, SMEM_BYTES, stream>>>(ctx, lens, W1, qb, wv, partial);
    finalize_kernel<<<B_, DV_, 0, stream>>>(partial, out);
}
